// TransactionGraphGAT_61108794687724
// MI455X (gfx1250) — compile-verified, matrix operand order checked by reading, not run on hardware
//
#include <hip/hip_runtime.h>
#include <hip/hip_bf16.h>
#include <math.h>

typedef __attribute__((ext_vector_type(16))) __bf16 v16bf;
typedef __attribute__((ext_vector_type(8)))  float  v8f;

#define DH 128
#define HEADS 4
#define CH 32

__device__ __forceinline__ float gelu_exact(float x) {
    return 0.5f * x * (1.0f + erff(x * 0.70710678118654752f));
}
__device__ __forceinline__ float leaky02(float x) {
    return x > 0.0f ? x : 0.2f * x;
}
// float atomic max via int punning (works for mixed signs; init = -inf)
__device__ __forceinline__ void atomicMaxFloat(float* addr, float v) {
    if (v >= 0.0f) atomicMax((int*)addr, __float_as_int(v));
    else           atomicMin((unsigned int*)addr, __float_as_uint(v));
}

__device__ __forceinline__ v16bf frag_from4(float4 r0, float4 r1, float4 r2, float4 r3) {
    v16bf a;
    a[0]  = (__bf16)r0.x; a[1]  = (__bf16)r0.y; a[2]  = (__bf16)r0.z; a[3]  = (__bf16)r0.w;
    a[4]  = (__bf16)r1.x; a[5]  = (__bf16)r1.y; a[6]  = (__bf16)r1.z; a[7]  = (__bf16)r1.w;
    a[8]  = (__bf16)r2.x; a[9]  = (__bf16)r2.y; a[10] = (__bf16)r2.z; a[11] = (__bf16)r2.w;
    a[12] = (__bf16)r3.x; a[13] = (__bf16)r3.y; a[14] = (__bf16)r3.z; a[15] = (__bf16)r3.w;
    return a;
}

// ---------- pack W[K,128] f32 (row-major) into WMMA B-fragment order, bf16 ----------
// layout: [kb][col][half][16], so a lane's whole fragment = contiguous 32B
__global__ void pack_w_kernel(const float* __restrict__ W, __bf16* __restrict__ Wp, int K) {
    const int idx = blockIdx.x * blockDim.x + threadIdx.x;
    const int total = (K >> 5) * 256;           // kb * (128 cols * 2 halves)
    if (idx >= total) return;
    const int half = idx & 1;
    const int col  = (idx >> 1) & 127;
    const int kb   = idx >> 8;
    __bf16* o = Wp + (size_t)idx * 16;
    const int k0 = kb * 32 + half * 8;
#pragma unroll
    for (int j = 0; j < 8; ++j) o[j]     = (__bf16)W[(size_t)(k0 + j) * DH + col];
#pragma unroll
    for (int j = 0; j < 8; ++j) o[8 + j] = (__bf16)W[(size_t)(k0 + 16 + j) * DH + col];
}

// ---------------- WMMA GEMM: out[N,128] = act(A[N,K] @ W[K,128] + bias) ----------------
__global__ void gemm128_wmma(const float* __restrict__ A, const __bf16* __restrict__ Wp,
                             const float* __restrict__ bias, float* __restrict__ out,
                             int N, int K, int act) {
    const int wave = threadIdx.x >> 5;
    const int lane = threadIdx.x & 31;
    const int rowTile = blockIdx.x * 16;
    const int rlo  = lane & 15;
    const int half = lane >> 4;
    const int col  = wave * 16 + rlo;
    const int arow = min(rowTile + rlo, N - 1);

    v8f acc = {};
    for (int k0 = 0; k0 < K; k0 += 32) {
        const float* ap = A + (long long)arow * K + k0 + (half << 3);
        const float4 r0 = *(const float4*)(ap);
        const float4 r1 = *(const float4*)(ap + 4);
        const float4 r2 = *(const float4*)(ap + 16);
        const float4 r3 = *(const float4*)(ap + 20);
        const v16bf a = frag_from4(r0, r1, r2, r3);
        const v16bf b = *(const v16bf*)(Wp + ((((size_t)(k0 >> 5) * 128 + col) * 2 + half) << 4));
        acc = __builtin_amdgcn_wmma_f32_16x16x32_bf16(false, a, false, b,
                                                      (short)0, acc, false, false);
    }
#pragma unroll
    for (int v = 0; v < 8; ++v) {
        const int row = rowTile + v + (half ? 8 : 0);
        if (row < N) {
            float x = acc[v];
            if (bias) x += bias[col];
            if (act)  x = gelu_exact(x);
            out[(long long)row * DH + col] = x;
        }
    }
}

// ---------------- WMMA GEMM over combined features [h, enc, h*enc] (K=384) ----------------
__device__ __forceinline__ float4 comb_load4(const float* __restrict__ h,
                                             const float* __restrict__ enc,
                                             long long rowoff, int k) {
    if (k < DH)     return *(const float4*)(h + rowoff + k);
    if (k < 2 * DH) return *(const float4*)(enc + rowoff + (k - DH));
    const int kk = k - 2 * DH;
    const float4 a = *(const float4*)(h + rowoff + kk);
    const float4 b = *(const float4*)(enc + rowoff + kk);
    return make_float4(a.x * b.x, a.y * b.y, a.z * b.z, a.w * b.w);
}

__global__ void gemm_combined_wmma(const float* __restrict__ h, const float* __restrict__ enc,
                                   const __bf16* __restrict__ Wp, const float* __restrict__ bias,
                                   float* __restrict__ out, int N) {
    const int wave = threadIdx.x >> 5;
    const int lane = threadIdx.x & 31;
    const int rowTile = blockIdx.x * 16;
    const int rlo  = lane & 15;
    const int half = lane >> 4;
    const int col  = wave * 16 + rlo;
    const int arow = min(rowTile + rlo, N - 1);
    const long long rowoff = (long long)arow * DH;

    v8f acc = {};
    for (int k0 = 0; k0 < 3 * DH; k0 += 32) {
        const int kb = k0 + (half << 3);
        const float4 r0 = comb_load4(h, enc, rowoff, kb);
        const float4 r1 = comb_load4(h, enc, rowoff, kb + 4);
        const float4 r2 = comb_load4(h, enc, rowoff, kb + 16);
        const float4 r3 = comb_load4(h, enc, rowoff, kb + 20);
        const v16bf a = frag_from4(r0, r1, r2, r3);
        const v16bf b = *(const v16bf*)(Wp + ((((size_t)(k0 >> 5) * 128 + col) * 2 + half) << 4));
        acc = __builtin_amdgcn_wmma_f32_16x16x32_bf16(false, a, false, b,
                                                      (short)0, acc, false, false);
    }
#pragma unroll
    for (int v = 0; v < 8; ++v) {
        const int row = rowTile + v + (half ? 8 : 0);
        if (row < N) out[(long long)row * DH + col] = gelu_exact(acc[v] + bias[col]);
    }
}

// ---------------- per-(node,head): alpha_src/alpha_dst dots + init m/den ----------------
__global__ void alpha_init_kernel(const float* __restrict__ xw,
                                  const float* __restrict__ a_src, const float* __restrict__ a_dst,
                                  float* __restrict__ al, float* __restrict__ ar,
                                  float* __restrict__ m, float* __restrict__ den, int N) {
    const int idx = blockIdx.x * blockDim.x + threadIdx.x;
    if (idx >= N * HEADS) return;
    const int n = idx >> 2, h = idx & 3;
    const float* row = xw + (long long)n * DH + h * CH;
    float sa = 0.f, sd = 0.f;
#pragma unroll
    for (int c = 0; c < CH; ++c) {
        const float v = row[c];
        sa += v * a_src[h * CH + c];
        sd += v * a_dst[h * CH + c];
    }
    al[idx] = sa; ar[idx] = sd;
    m[idx] = -INFINITY; den[idx] = 0.0f;
}

__global__ void zero_kernel(float* __restrict__ p, long long n) {
    const long long i = (long long)blockIdx.x * blockDim.x + threadIdx.x;
    if (i < n) p[i] = 0.0f;
}

// ---------------- edge passes ----------------
__device__ __forceinline__ void edge_nodes(const long long* __restrict__ ei, long long E,
                                           long long e, int& s, int& d) {
    if (e < E) { s = (int)ei[e]; d = (int)ei[E + e]; }
    else       { s = d = (int)(e - E); }
}

__global__ void edge_max_kernel(const long long* __restrict__ ei, long long E, long long Etot,
                                const float* __restrict__ al, const float* __restrict__ ar,
                                float* __restrict__ m) {
    const long long idx = (long long)blockIdx.x * blockDim.x + threadIdx.x;
    if (idx >= Etot * HEADS) return;
    const long long e = idx >> 2; const int h = (int)(idx & 3);
    int s, d; edge_nodes(ei, E, e, s, d);
    const float ev = leaky02(al[s * HEADS + h] + ar[d * HEADS + h]);
    atomicMaxFloat(&m[d * HEADS + h], ev);
}

__global__ void edge_expsum_kernel(const long long* __restrict__ ei, long long E, long long Etot,
                                   const float* __restrict__ al, const float* __restrict__ ar,
                                   const float* __restrict__ m, float* __restrict__ wbuf,
                                   float* __restrict__ den) {
    const long long idx = (long long)blockIdx.x * blockDim.x + threadIdx.x;
    if (idx >= Etot * HEADS) return;
    const long long e = idx >> 2; const int h = (int)(idx & 3);
    int s, d; edge_nodes(ei, E, e, s, d);
    const float ev = leaky02(al[s * HEADS + h] + ar[d * HEADS + h]);
    const float w = expf(ev - m[d * HEADS + h]);
    wbuf[idx] = w;
    atomicAdd(&den[d * HEADS + h], w);
}

// one thread per (edge, 4-channel group): float4 gather + 4 atomic f32 adds
__global__ void edge_scatter_kernel(const long long* __restrict__ ei, long long E, long long Etot,
                                    const float* __restrict__ xw, const float* __restrict__ wbuf,
                                    const float* __restrict__ den, float* __restrict__ out) {
    const long long idx = (long long)blockIdx.x * blockDim.x + threadIdx.x;
    if (idx >= Etot * (DH / 4)) return;
    const long long e = idx >> 5;
    const int q  = (int)(idx & 31);
    const int dc = q << 2;
    const int h  = q >> 3;
    int s, d; edge_nodes(ei, E, e, s, d);
    const float a = wbuf[e * HEADS + h] / (den[d * HEADS + h] + 1e-16f);
    const float4 x = *(const float4*)(xw + (long long)s * DH + dc);
    float* o = out + (long long)d * DH + dc;
    atomicAdd(o + 0, x.x * a);
    atomicAdd(o + 1, x.y * a);
    atomicAdd(o + 2, x.z * a);
    atomicAdd(o + 3, x.w * a);
}

// ---------------- fused (acc + gat_bias + enc) -> layernorm -> gelu (wave per node) ----------------
__global__ void ln_gelu_kernel(const float* __restrict__ acc, const float* __restrict__ gbias,
                               const float* __restrict__ enc, const float* __restrict__ g,
                               const float* __restrict__ be, float* __restrict__ out, int N) {
    const int wave = threadIdx.x >> 5, lane = threadIdx.x & 31;
    const int n = blockIdx.x * (blockDim.x >> 5) + wave;
    if (n >= N) return;
    float x[4]; float s = 0.f, s2 = 0.f;
#pragma unroll
    for (int i = 0; i < 4; ++i) {
        const int c = i * 32 + lane;
        const float v = acc[(long long)n * DH + c] + gbias[c] + enc[(long long)n * DH + c];
        x[i] = v; s += v; s2 += v * v;
    }
#pragma unroll
    for (int msk = 16; msk >= 1; msk >>= 1) {
        s  += __shfl_xor(s,  msk, 32);
        s2 += __shfl_xor(s2, msk, 32);
    }
    const float mean = s * (1.0f / DH);
    const float var  = s2 * (1.0f / DH) - mean * mean;
    const float rs   = rsqrtf(var + 1e-5f);
#pragma unroll
    for (int i = 0; i < 4; ++i) {
        const int c = i * 32 + lane;
        const float y = (x[i] - mean) * rs * g[c] + be[c];
        out[(long long)n * DH + c] = gelu_exact(y);
    }
}

// ---------------- final projection hid[N,128] @ Wc2[128,2] + bc2 (wave per node) ----------------
__global__ void final_out_kernel(const float* __restrict__ hid, const float* __restrict__ Wc2,
                                 const float* __restrict__ bc2, float* __restrict__ out, int N) {
    const int wave = threadIdx.x >> 5, lane = threadIdx.x & 31;
    const int n = blockIdx.x * (blockDim.x >> 5) + wave;
    if (n >= N) return;
    float s0 = 0.f, s1 = 0.f;
#pragma unroll
    for (int i = 0; i < 4; ++i) {
        const int c = i * 32 + lane;
        const float hv = hid[(long long)n * DH + c];
        s0 += hv * Wc2[c * 2 + 0];
        s1 += hv * Wc2[c * 2 + 1];
    }
#pragma unroll
    for (int msk = 16; msk >= 1; msk >>= 1) {
        s0 += __shfl_xor(s0, msk, 32);
        s1 += __shfl_xor(s1, msk, 32);
    }
    if (lane == 0) {
        out[(long long)n * 2 + 0] = s0 + bc2[0];
        out[(long long)n * 2 + 1] = s1 + bc2[1];
    }
}

extern "C" void kernel_launch(void* const* d_in, const int* in_sizes, int n_in,
                              void* d_out, int out_size, void* d_ws, size_t ws_size,
                              hipStream_t stream) {
    const float* inputs = (const float*)d_in[0];
    const float* W1  = (const float*)d_in[1];
    const float* b1  = (const float*)d_in[2];
    const float* Wg1 = (const float*)d_in[3];
    const float* as1 = (const float*)d_in[4];
    const float* ad1 = (const float*)d_in[5];
    const float* bg1 = (const float*)d_in[6];
    const float* g1  = (const float*)d_in[7];
    const float* be1 = (const float*)d_in[8];
    const float* Wg2 = (const float*)d_in[9];
    const float* as2 = (const float*)d_in[10];
    const float* ad2 = (const float*)d_in[11];
    const float* bg2 = (const float*)d_in[12];
    const float* g2  = (const float*)d_in[13];
    const float* be2 = (const float*)d_in[14];
    const float* Wc1 = (const float*)d_in[15];
    const float* bc1 = (const float*)d_in[16];
    const float* Wc2 = (const float*)d_in[17];
    const float* bc2 = (const float*)d_in[18];
    const long long* ei = (const long long*)d_in[19];

    const int  N    = in_sizes[0] / 64;
    const long long E    = (long long)in_sizes[19] / 2;
    const long long Etot = E + N;

    // ---- workspace layout ----
    float* A  = (float*)d_ws;                     // enc           N*128
    float* B  = A  + (long long)N * DH;           // ping          N*128
    float* Cb = B  + (long long)N * DH;           // pong          N*128
    float* al = Cb + (long long)N * DH;           // N*4
    float* ar = al + (long long)N * HEADS;
    float* mB = ar + (long long)N * HEADS;
    float* dn = mB + (long long)N * HEADS;
    float* wb = dn + (long long)N * HEADS;        // Etot*4
    size_t off = (size_t)((char*)(wb + Etot * HEADS) - (char*)d_ws);
    off = (off + 63) & ~(size_t)63;
    __bf16* W1p  = (__bf16*)((char*)d_ws + off); off += (size_t)(64  >> 5) * 4096 * sizeof(__bf16);
    __bf16* Wg1p = (__bf16*)((char*)d_ws + off); off += (size_t)(128 >> 5) * 4096 * sizeof(__bf16);
    __bf16* Wg2p = (__bf16*)((char*)d_ws + off); off += (size_t)(128 >> 5) * 4096 * sizeof(__bf16);
    __bf16* Wc1p = (__bf16*)((char*)d_ws + off); off += (size_t)(384 >> 5) * 4096 * sizeof(__bf16);

    const int gemmGrid = (N + 15) / 16;
    const int nhGrid   = (N * HEADS + 255) / 256;
    const int zGrid    = (int)(((long long)N * DH + 255) / 256);
    const int e4Grid   = (int)((Etot * HEADS + 255) / 256);
    const int e32Grid  = (int)((Etot * (DH / 4) + 255) / 256);
    const int nodeGrid = (N + 7) / 8;   // 8 waves/block, wave per node

    // pack all weight matrices into bf16 WMMA-fragment order (tiny, reused by every tile)
    pack_w_kernel<<<((64  >> 5) * 256 + 255) / 256, 256, 0, stream>>>(W1,  W1p,  64);
    pack_w_kernel<<<((128 >> 5) * 256 + 255) / 256, 256, 0, stream>>>(Wg1, Wg1p, 128);
    pack_w_kernel<<<((128 >> 5) * 256 + 255) / 256, 256, 0, stream>>>(Wg2, Wg2p, 128);
    pack_w_kernel<<<((384 >> 5) * 256 + 255) / 256, 256, 0, stream>>>(Wc1, Wc1p, 384);

    // encoder: enc = gelu(inputs @ W1 + b1)
    gemm128_wmma<<<gemmGrid, 256, 0, stream>>>(inputs, W1p, b1, A, N, 64, 1);

    // ---- GAT layer 1: xw -> B, accumulate -> Cb, h1 -> B ----
    gemm128_wmma<<<gemmGrid, 256, 0, stream>>>(A, Wg1p, nullptr, B, N, DH, 0);
    alpha_init_kernel<<<nhGrid, 256, 0, stream>>>(B, as1, ad1, al, ar, mB, dn, N);
    zero_kernel<<<zGrid, 256, 0, stream>>>(Cb, (long long)N * DH);
    edge_max_kernel<<<e4Grid, 256, 0, stream>>>(ei, E, Etot, al, ar, mB);
    edge_expsum_kernel<<<e4Grid, 256, 0, stream>>>(ei, E, Etot, al, ar, mB, wb, dn);
    edge_scatter_kernel<<<e32Grid, 256, 0, stream>>>(ei, E, Etot, B, wb, dn, Cb);
    ln_gelu_kernel<<<nodeGrid, 256, 0, stream>>>(Cb, bg1, A, g1, be1, B, N);   // h1 -> B

    // ---- GAT layer 2: xw -> Cb, accumulate -> B, h2 -> Cb ----
    gemm128_wmma<<<gemmGrid, 256, 0, stream>>>(B, Wg2p, nullptr, Cb, N, DH, 0);
    alpha_init_kernel<<<nhGrid, 256, 0, stream>>>(Cb, as2, ad2, al, ar, mB, dn, N);
    zero_kernel<<<zGrid, 256, 0, stream>>>(B, (long long)N * DH);
    edge_max_kernel<<<e4Grid, 256, 0, stream>>>(ei, E, Etot, al, ar, mB);
    edge_expsum_kernel<<<e4Grid, 256, 0, stream>>>(ei, E, Etot, al, ar, mB, wb, dn);
    edge_scatter_kernel<<<e32Grid, 256, 0, stream>>>(ei, E, Etot, Cb, wb, dn, B);
    ln_gelu_kernel<<<nodeGrid, 256, 0, stream>>>(B, bg2, A, g2, be2, Cb, N);   // h2 -> Cb

    // ---- classifier: hid = gelu([h2, enc, h2*enc] @ Wc1 + bc1) -> B; out = hid @ Wc2 + bc2 ----
    gemm_combined_wmma<<<gemmGrid, 256, 0, stream>>>(Cb, A, Wc1p, bc1, B, N);
    final_out_kernel<<<nodeGrid, 256, 0, stream>>>(B, Wc2, bc2, (float*)d_out, N);
}
